// PEneoDecoder_84293028151981
// MI455X (gfx1250) — matrix operand order repaired and checked by reading, not compiled, not run on hardware
//
#include <hip/hip_runtime.h>

// ---------------------------------------------------------------------------
// PEneo decoder head, restructured:
//   h1 = relu(X @ W1 + b1)                       (1024 x 768)
//   h2 = relu(h1 @ W2 + b2)                      (1024 x 384)
//   L  = h2 @ combine_w[:384]                    (1024 x 384)
//   R  = h2 @ combine_w[384:]                    (1024 x 384)
//   per pair (i<=j): out = tanh(L[i]+R[j]+cb) @ Wh + bh   (384 -> 14, padded 16)
// All GEMMs via v_wmma_f32_16x16x32_f16 (f32 accumulate), LDS-staged tiles.
// ---------------------------------------------------------------------------

typedef __attribute__((ext_vector_type(16))) _Float16 v16h;
typedef __attribute__((ext_vector_type(8)))  float    v8f;

#define S_LEN  512
#define HD     384
#define NPAIR  131328     // S*(S+1)/2
#define NHEAD  14

union FragU { uint4 u[2]; v16h v; };
union H4U   { unsigned long long u; _Float16 h[4]; };

__device__ __forceinline__ void store_h4(_Float16* dst, float4 f) {
    H4U h4;
    h4.h[0] = (_Float16)f.x; h4.h[1] = (_Float16)f.y;
    h4.h[2] = (_Float16)f.z; h4.h[3] = (_Float16)f.w;
    *(unsigned long long*)dst = h4.u;   // 8B-aligned
}

// ---------------------------------------------------------------------------
// LDS-staged GEMM: 128 threads (4 waves), 64x64 output tile per block.
// A (MxK) row-major f32, B (KxN) row-major f32, C f32, optional bias+relu.
// M % 64 == 0, N % 64 == 0, K % 32 == 0.
//
// WMMA 16-bit A 16x32 lane layout used by this kernel (contiguous K):
//   lanes 0-15 : row m, K = k0+0..15
//   lanes16-31 : row m, K = k0+16..31
// B 32x16: lanes 0-15 col n hold K=k0+0..15; lanes 16-31 hold K=k0+16..31.
// C/D: VGPR v -> row v (lanes 0-15) / row v+8 (lanes 16-31), col = lane&15.
// ---------------------------------------------------------------------------
__global__ void __launch_bounds__(128)
gemm_wmma_f16(const float* __restrict__ A, const float* __restrict__ Bm,
              const float* __restrict__ bias, float* __restrict__ C,
              int M, int N, int K, int relu) {
    __shared__ __align__(16) _Float16 sA [64 * 32];   // row-major, stride 32
    __shared__ __align__(16) _Float16 sBT[64 * 32];   // n-major,  stride 32

    const int tid  = threadIdx.x;          // 0..127
    const int wv   = tid >> 5;             // wave 0..3
    const int lane = tid & 31;
    const int row0 = blockIdx.y * 64;
    const int col0 = blockIdx.x * 64;
    const int m    = lane & 15;
    const int aoff = (lane < 16) ? 0 : 16;  // A K-offset within 32-chunk
    const int boff = (lane < 16) ? 0 : 16;  // B K-offset within 32-chunk

    v8f acc[4];
#pragma unroll
    for (int t = 0; t < 4; ++t) acc[t] = (v8f){0.f,0.f,0.f,0.f,0.f,0.f,0.f,0.f};

    for (int k0 = 0; k0 < K; k0 += 32) {
        // ---- stage A tile 64x32: 512 float4 loads across 128 threads
#pragma unroll
        for (int q = 0; q < 4; ++q) {
            int idx = tid + 128 * q;            // 0..511
            int ar  = idx >> 3;                 // 0..63
            int as  = idx & 7;                  // float4 segment 0..7
            float4 f = *(const float4*)(A + (size_t)(row0 + ar) * K + k0 + as * 4);
            store_h4(&sA[ar * 32 + as * 4], f);
        }
        // ---- stage B tile 32x64 transposed into sBT[n][k]
#pragma unroll
        for (int q = 0; q < 4; ++q) {
            int idx = tid + 128 * q;            // 0..511
            int br  = idx >> 4;                 // K row 0..31
            int bs  = idx & 15;                 // float4 segment 0..15
            float4 f = *(const float4*)(Bm + (size_t)(k0 + br) * N + col0 + bs * 4);
            int n = bs * 4;
            sBT[(n + 0) * 32 + br] = (_Float16)f.x;
            sBT[(n + 1) * 32 + br] = (_Float16)f.y;
            sBT[(n + 2) * 32 + br] = (_Float16)f.z;
            sBT[(n + 3) * 32 + br] = (_Float16)f.w;
        }
        // ---- prefetch next K tile (global_prefetch_b8)
        if (k0 + 32 < K) {
            __builtin_prefetch(A + (size_t)(row0 + (tid >> 1)) * K + k0 + 32 + (tid & 1) * 16, 0, 1);
            __builtin_prefetch(Bm + (size_t)(k0 + 32 + (tid >> 2)) * N + col0 + (tid & 3) * 16, 0, 1);
        }
        __syncthreads();

        // ---- each wave: 16 rows x 64 cols
        const int mrow = wv * 16 + m;
        FragU a;
        a.u[0] = *(const uint4*)&sA[mrow * 32 + aoff];        // K aoff..aoff+7
        a.u[1] = *(const uint4*)&sA[mrow * 32 + 8 + aoff];    // K 8+aoff..
#pragma unroll
        for (int t = 0; t < 4; ++t) {
            const int n = t * 16 + m;
            FragU b;
            b.u[0] = *(const uint4*)&sBT[n * 32 + boff];      // K boff..boff+7
            b.u[1] = *(const uint4*)&sBT[n * 32 + boff + 8];  // K boff+8..+15
            acc[t] = __builtin_amdgcn_wmma_f32_16x16x32_f16(
                false, a.v, false, b.v, (short)0, acc[t], false, false);
        }
        __syncthreads();
    }

    const int mbase = row0 + wv * 16 + ((lane < 16) ? 0 : 8);
#pragma unroll
    for (int t = 0; t < 4; ++t) {
        const int n = col0 + t * 16 + m;
        const float bv = bias ? bias[n] : 0.f;
#pragma unroll
        for (int v = 0; v < 8; ++v) {
            float val = acc[t][v] + bv;
            if (relu) val = fmaxf(val, 0.f);
            C[(size_t)(mbase + v) * N + n] = val;
        }
    }
}

// ---------------------------------------------------------------------------
// Pack 5 head weights (384x{2,3,3,3,3}) into transposed f16 WhT[16][384]
// (cols 14,15 zero) and bias bh[16].
// ---------------------------------------------------------------------------
__global__ void pack_heads(const float* __restrict__ le_w,  const float* __restrict__ elh_w,
                           const float* __restrict__ elt_w, const float* __restrict__ lgh_w,
                           const float* __restrict__ lgt_w,
                           const float* __restrict__ le_b,  const float* __restrict__ elh_b,
                           const float* __restrict__ elt_b, const float* __restrict__ lgh_b,
                           const float* __restrict__ lgt_b,
                           _Float16* __restrict__ WhT, float* __restrict__ bh) {
    int idx = blockIdx.x * blockDim.x + threadIdx.x;
    if (idx < 16 * HD) {
        int n = idx / HD;
        int k = idx - n * HD;
        float v = 0.f;
        if      (n <  2) v = le_w [k * 2 +  n      ];
        else if (n <  5) v = elh_w[k * 3 + (n -  2)];
        else if (n <  8) v = elt_w[k * 3 + (n -  5)];
        else if (n < 11) v = lgh_w[k * 3 + (n -  8)];
        else if (n < 14) v = lgt_w[k * 3 + (n - 11)];
        WhT[idx] = (_Float16)v;
    }
    if (idx < 16) {
        int n = idx;
        float v = 0.f;
        if      (n <  2) v = le_b [n];
        else if (n <  5) v = elh_b[n -  2];
        else if (n <  8) v = elt_b[n -  5];
        else if (n < 11) v = lgh_b[n -  8];
        else if (n < 14) v = lgt_b[n - 11];
        bh[n] = v;
    }
}

// ---------------------------------------------------------------------------
// Pair kernel: 128 threads / block, 64 pair-rows / block (16 per wave).
//   1) invert triangular index p -> (i,j)   (rows 0..63)
//   2) all threads: sA[64][384] = f16(tanh(L[i]+R[j]+cb))  (coalesced, LDS)
//   3) each wave: 12x wmma_f32_16x16x32_f16 against LDS head weights
// 2*NPAIR = 262656 rows, divisible by 64 -> no partial tiles.
// ---------------------------------------------------------------------------
__global__ void __launch_bounds__(128)
pair_head_kernel(const float* __restrict__ L, const float* __restrict__ R,
                 const float* __restrict__ cb,
                 const _Float16* __restrict__ WhT, const float* __restrict__ bh,
                 float* __restrict__ out) {
    __shared__ int sLbase[64];
    __shared__ int sRbase[64];
    __shared__ __align__(16) _Float16 sA[64 * HD];   // 48 KB tanh tile
    __shared__ __align__(16) _Float16 sB[16 * HD];   // 12 KB head weights (n-major)

    const int tid = threadIdx.x;        // 0..127
    const int r0  = blockIdx.x * 64;    // global pair-row base

    if (tid < 64) {
        int r = r0 + tid;
        int b = r / NPAIR;
        int p = r - b * NPAIR;
        // invert start(i) = i*S - i*(i-1)/2 <= p  via sqrt + fixup
        float tS = 2.0f * S_LEN + 1.0f;                       // 1025
        float disc = fmaxf(tS * tS - 8.0f * (float)p, 0.f);
        int i = (int)((tS - sqrtf(disc)) * 0.5f);
        if (i < 0) i = 0;
        if (i > S_LEN - 1) i = S_LEN - 1;
        while (i > 0 && (i * S_LEN - (i * (i - 1)) / 2) > p) --i;
        while (i < S_LEN - 1 && ((i + 1) * S_LEN - ((i + 1) * i) / 2) <= p) ++i;
        int start = i * S_LEN - (i * (i - 1)) / 2;
        int j = i + (p - start);
        sLbase[tid] = (b * S_LEN + i) * HD;
        sRbase[tid] = (b * S_LEN + j) * HD;
    }
    for (int idx = tid; idx < 16 * HD; idx += 128) sB[idx] = WhT[idx];

    // combine-bias: 3 fixed columns per thread
    const int c0 = tid, c1 = tid + 128, c2 = tid + 256;
    const float cb0 = cb[c0], cb1 = cb[c1], cb2 = cb[c2];
    __syncthreads();

    for (int r = 0; r < 64; ++r) {
        const int lb = sLbase[r], rb = sRbase[r];
        float v0 = L[lb + c0] + R[rb + c0] + cb0;
        float v1 = L[lb + c1] + R[rb + c1] + cb1;
        float v2 = L[lb + c2] + R[rb + c2] + cb2;
        sA[r * HD + c0] = (_Float16)tanhf(v0);
        sA[r * HD + c1] = (_Float16)tanhf(v1);
        sA[r * HD + c2] = (_Float16)tanhf(v2);
    }
    __syncthreads();

    // each wave multiplies its own 16x384 strip by the 384x16 head matrix
    const int wv   = tid >> 5;
    const int lane = tid & 31;
    const int m    = lane & 15;
    const int aoff = (lane < 16) ? 0 : 16;
    const int boff = (lane < 16) ? 0 : 16;
    const int mrow = wv * 16 + m;
    v8f acc = (v8f){0.f,0.f,0.f,0.f,0.f,0.f,0.f,0.f};
#pragma unroll
    for (int kt = 0; kt < HD / 32; ++kt) {
        const int k0 = kt * 32;
        FragU a, b;
        a.u[0] = *(const uint4*)&sA[mrow * HD + k0 + aoff];
        a.u[1] = *(const uint4*)&sA[mrow * HD + k0 + aoff + 8];
        b.u[0] = *(const uint4*)&sB[m * HD + k0 + boff];
        b.u[1] = *(const uint4*)&sB[m * HD + k0 + boff + 8];
        acc = __builtin_amdgcn_wmma_f32_16x16x32_f16(
            false, a.v, false, b.v, (short)0, acc, false, false);
    }
    const int n = m;
    if (n < NHEAD) {
        const float bv    = bh[n];
        const int   rbase = r0 + wv * 16 + ((lane < 16) ? 0 : 8);
#pragma unroll
        for (int v = 0; v < 8; ++v)
            out[(size_t)(rbase + v) * NHEAD + n] = acc[v] + bv;
    }
}

// ---------------------------------------------------------------------------
extern "C" void kernel_launch(void* const* d_in, const int* in_sizes, int n_in,
                              void* d_out, int out_size, void* d_ws, size_t ws_size,
                              hipStream_t stream) {
    const float* seq   = (const float*)d_in[0];   // (2,512,768)
    const float* w1    = (const float*)d_in[1];   // (768,768)
    const float* b1    = (const float*)d_in[2];   // (768)
    const float* w2    = (const float*)d_in[3];   // (768,384)
    const float* b2    = (const float*)d_in[4];   // (384)
    const float* cw    = (const float*)d_in[5];   // (768,384)
    const float* cb    = (const float*)d_in[6];   // (384)
    const float* le_w  = (const float*)d_in[7];
    const float* le_b  = (const float*)d_in[8];
    const float* elh_w = (const float*)d_in[9];
    const float* elh_b = (const float*)d_in[10];
    const float* elt_w = (const float*)d_in[11];
    const float* elt_b = (const float*)d_in[12];
    const float* lgh_w = (const float*)d_in[13];
    const float* lgh_b = (const float*)d_in[14];
    const float* lgt_w = (const float*)d_in[15];
    const float* lgt_b = (const float*)d_in[16];
    float* out = (float*)d_out;

    // workspace layout (~7.9 MB)
    char* ws = (char*)d_ws;
    float*    h1  = (float*)(ws);                                   // 1024*768
    float*    h2  = (float*)(ws + (size_t)1024 * 768 * 4);          // 1024*384
    float*    Lm  = h2 + (size_t)1024 * 384;                        // 1024*384
    float*    Rm  = Lm + (size_t)1024 * 384;                        // 1024*384
    _Float16* WhT = (_Float16*)(Rm + (size_t)1024 * 384);           // 16*384 f16
    float*    bh  = (float*)((char*)WhT + 16 * HD * 2);             // 16 f32

    pack_heads<<<dim3((16 * HD + 255) / 256), 256, 0, stream>>>(
        le_w, elh_w, elt_w, lgh_w, lgt_w,
        le_b, elh_b, elt_b, lgh_b, lgt_b, WhT, bh);

    // h1 = relu(seq @ w1 + b1)
    gemm_wmma_f16<<<dim3(768 / 64, 1024 / 64), 128, 0, stream>>>(
        seq, w1, b1, h1, 1024, 768, 768, 1);
    // h2 = relu(h1 @ w2 + b2)
    gemm_wmma_f16<<<dim3(384 / 64, 1024 / 64), 128, 0, stream>>>(
        h1, w2, b2, h2, 1024, 384, 768, 1);
    // L = h2 @ w_top  (combine_w rows 0..383)
    gemm_wmma_f16<<<dim3(384 / 64, 1024 / 64), 128, 0, stream>>>(
        h2, cw, nullptr, Lm, 1024, 384, 384, 0);
    // R = h2 @ w_bot  (combine_w rows 384..767)
    gemm_wmma_f16<<<dim3(384 / 64, 1024 / 64), 128, 0, stream>>>(
        h2, cw + (size_t)384 * 384, nullptr, Rm, 1024, 384, 384, 0);

    // fused tanh + 5-head projection over all 2*131328 pairs
    pair_head_kernel<<<(2 * NPAIR) / 64, 128, 0, stream>>>(Lm, Rm, cb, WhT, bh, out);
}
